// GAFM_14937896255494
// MI455X (gfx1250) — matrix operand order repaired and checked by reading, not compiled, not run on hardware
//
#include <hip/hip_runtime.h>
#include <hip/hip_bf16.h>

typedef _Float16 v16h __attribute__((ext_vector_type(16)));
typedef _Float16 half8 __attribute__((ext_vector_type(8)));
typedef float    v8f  __attribute__((ext_vector_type(8)));
typedef float    f4   __attribute__((ext_vector_type(4)));

#define BTOT   65536
#define KNB    5
#define DIM    64
#define H1DIM  100
#define H2DIM  1024
#define H3DIM  512
#define MT     16     // row tile for stage2
#define MT2    32     // row tile for MLP kernel (2 M-subtiles of 16)

// ---------------- WMMA helpers (contiguous k order, same in A and B) ----------------

__device__ __forceinline__ v8f wmma32(v16h a, v16h b, v8f c) {
  // D = A(16x32 f16) * B(32x16 f16) + C(16x16 f32)
  return __builtin_amdgcn_wmma_f32_16x16x32_f16(false, a, false, b, (short)0, c,
                                                false, false);
}

// A-frag: 16x32 f16 from an LDS tile (row-major, 'stride' f16 per row).
// lane<16 : row=lane,    k = kbase+{0..15}
// lane>=16: row=lane-16, k = kbase+{16..31}
__device__ __forceinline__ v16h ldA(const _Float16* lds, int kbase, int stride) {
  const int lane = threadIdx.x & 31;
  const int m = lane & 15;
  const int h = lane >> 4;
  const _Float16* p = lds + m * stride + kbase + h * 16;
  half8 lo = *(const half8*)(p);
  half8 hi = *(const half8*)(p + 8);
  v16h a;
#pragma unroll
  for (int i = 0; i < 8; ++i) { a[i] = lo[i]; a[8 + i] = hi[i]; }
  return a;
}

// B-frag: 32x16 f16, B[k][n] = W[nbase+n][kbase+k], W row-major [N][Kfull] f16.
// lane<16 : col=lane, k = kbase+0..15 ; lane>=16: col=lane-16, k = kbase+16..31
__device__ __forceinline__ v16h ldB(const _Float16* W, int nbase, int kbase, int Kfull) {
  const int lane = threadIdx.x & 31;
  const int n = lane & 15;
  const int kh = lane >> 4;
  const _Float16* p = W + (size_t)(nbase + n) * Kfull + kbase + kh * 16;
  return *(const v16h*)(p);   // 32B contiguous
}

// ---------------- weight f32 -> f16 conversion ----------------

__global__ void cvt_f32_f16(const float* __restrict__ s, _Float16* __restrict__ d, int n) {
  int i = blockIdx.x * 256 + threadIdx.x;
  if (i < n) d[i] = (_Float16)s[i];
}

// ---------------- Stage 2: gathers + QKV linears (WMMA) + gating -> uv f16 ----

__global__ __launch_bounds__(128) void stage2_kernel(
    const float* __restrict__ items_emb, const float* __restrict__ df_emb,
    const float* __restrict__ pos_emb,   const float* __restrict__ neg_emb,
    const _Float16* __restrict__ WqH,  const _Float16* __restrict__ Wk1H,
    const _Float16* __restrict__ Wv1H, const _Float16* __restrict__ Wk2H,
    const _Float16* __restrict__ Wv2H,
    const float* __restrict__ bq,  const float* __restrict__ bk1,
    const float* __restrict__ bv1, const float* __restrict__ bk2,
    const float* __restrict__ bv2,
    const float* __restrict__ Wf1, const float* __restrict__ bf1,
    const float* __restrict__ Wf2, const float* __restrict__ bf2,
    const float* __restrict__ Wf3, const float* __restrict__ bf3,
    const float* __restrict__ Wf4, const float* __restrict__ bf4,
    const float* __restrict__ w2,  const float* __restrict__ w3,
    const int* __restrict__ iidx,
    const int* __restrict__ idx1, const int* __restrict__ nb1,
    const int* __restrict__ idx2, const int* __restrict__ nb2,
    const int* __restrict__ idx3, const int* __restrict__ nb3,
    _Float16* __restrict__ uvg) {

  __shared__ __align__(32) _Float16 Adf[MT][DIM];   // f16 A-tiles for WMMA
  __shared__ __align__(32) _Float16 Apos[MT][DIM];
  __shared__ __align__(32) _Float16 Aneg[MT][DIM];
  __shared__ __align__(16) float    Itm[MT][DIM];
  __shared__ __align__(16) float    Outs[5][MT][DIM]; // Q, pk, pv, nk, nv
  __shared__ float Hg[2][MT][H1DIM];
  __shared__ float Sg[2][MT];

  const int rowbase = blockIdx.x * MT;
  const int t   = threadIdx.x;
  const int r   = t >> 3;          // row in tile, 0..15
  const int off = (t & 7) * 8;     // 8 dims per thread
  const int b   = rowbase + r;

  // ---- gathers: FM aggregator (G1) ----
  {
    float s0[8], q0[8];
#pragma unroll
    for (int j = 0; j < 8; ++j) { s0[j] = 0.f; q0[j] = 0.f; }
#pragma unroll
    for (int k = 0; k < KNB; ++k) {
      int e = nb1[b * KNB + k];
      const f4* p = (const f4*)(df_emb + (size_t)e * DIM + off);
      f4 x = p[0], y = p[1];
      float v[8] = {x[0], x[1], x[2], x[3], y[0], y[1], y[2], y[3]};
#pragma unroll
      for (int j = 0; j < 8; ++j) { s0[j] += v[j]; q0[j] += v[j] * v[j]; }
    }
    int e = idx1[b];
    const f4* p = (const f4*)(df_emb + (size_t)e * DIM + off);
    f4 x = p[0], y = p[1];
    float tv[8] = {x[0], x[1], x[2], x[3], y[0], y[1], y[2], y[3]};
#pragma unroll
    for (int j = 0; j < 8; ++j)
      Adf[r][off + j] = (_Float16)(s0[j] * s0[j] - q0[j] + tv[j]);
  }
  // ---- gathers: weighted sums (G2 positive, G3 negative) ----
  {
    float ap[8], an[8];
#pragma unroll
    for (int j = 0; j < 8; ++j) { ap[j] = 0.f; an[j] = 0.f; }
#pragma unroll
    for (int k = 0; k < KNB; ++k) {
      int e2 = nb2[b * KNB + k]; float wk2 = w2[b * KNB + k];
      const f4* p2 = (const f4*)(pos_emb + (size_t)e2 * DIM + off);
      f4 x2 = p2[0], y2 = p2[1];
      float v2[8] = {x2[0], x2[1], x2[2], x2[3], y2[0], y2[1], y2[2], y2[3]};
      int e3 = nb3[b * KNB + k]; float wk3 = w3[b * KNB + k];
      const f4* p3 = (const f4*)(neg_emb + (size_t)e3 * DIM + off);
      f4 x3 = p3[0], y3 = p3[1];
      float v3[8] = {x3[0], x3[1], x3[2], x3[3], y3[0], y3[1], y3[2], y3[3]};
#pragma unroll
      for (int j = 0; j < 8; ++j) { ap[j] += wk2 * v2[j]; an[j] += wk3 * v3[j]; }
    }
    int e2 = idx2[b];
    const f4* p2 = (const f4*)(pos_emb + (size_t)e2 * DIM + off);
    f4 x2 = p2[0], y2 = p2[1];
    float t2[8] = {x2[0], x2[1], x2[2], x2[3], y2[0], y2[1], y2[2], y2[3]};
    int e3 = idx3[b];
    const f4* p3 = (const f4*)(neg_emb + (size_t)e3 * DIM + off);
    f4 x3 = p3[0], y3 = p3[1];
    float t3[8] = {x3[0], x3[1], x3[2], x3[3], y3[0], y3[1], y3[2], y3[3]};
#pragma unroll
    for (int j = 0; j < 8; ++j) {
      Apos[r][off + j] = (_Float16)(ap[j] + t2[j]);
      Aneg[r][off + j] = (_Float16)(an[j] + t3[j]);
    }
  }
  // ---- items ----
  {
    int e = iidx[b];
    const f4* p = (const f4*)(items_emb + (size_t)e * DIM + off);
    f4 x = p[0], y = p[1];
    float tv[8] = {x[0], x[1], x[2], x[3], y[0], y[1], y[2], y[3]};
#pragma unroll
    for (int j = 0; j < 8; ++j) Itm[r][off + j] = tv[j];
  }
  __syncthreads();

  // ---- five 64x64 linears via WMMA: wave w owns N-tile w for all matrices ----
  {
    const int wave = t >> 5, lane = t & 31;
    const int nbase = wave * 16;
    const _Float16* Ap5[5]  = {&Adf[0][0], &Apos[0][0], &Apos[0][0], &Aneg[0][0], &Aneg[0][0]};
    const _Float16* Wh5[5]  = {WqH, Wk1H, Wv1H, Wk2H, Wv2H};
    const float*    bs5[5]  = {bq, bk1, bv1, bk2, bv2};
#pragma unroll
    for (int m = 0; m < 5; ++m) {
      v16h a0 = ldA(Ap5[m], 0, DIM);
      v16h a1 = ldA(Ap5[m], 32, DIM);
      v8f acc = {};
      acc = wmma32(a0, ldB(Wh5[m], nbase, 0, DIM), acc);
      acc = wmma32(a1, ldB(Wh5[m], nbase, 32, DIM), acc);
      const int n  = nbase + (lane & 15);
      const int mo = (lane >> 4) * 8;
      const float bv = bs5[m][n];
#pragma unroll
      for (int rr = 0; rr < 8; ++rr) Outs[m][mo + rr][n] = acc[rr] + bv;
    }
  }
  __syncthreads();

  // ---- gating: h = relu(Wf{1,3} @ (k*Q) + b), 100-wide (VALU path) ----
  for (int idx = t; idx < 2 * MT * H1DIM; idx += 128) {
    int branch = idx / (MT * H1DIM);
    int rem = idx - branch * (MT * H1DIM);
    int m = rem / H1DIM, o = rem - m * H1DIM;
    const float* W  = branch ? Wf3 : Wf1;
    const float* bb = branch ? bf3 : bf1;
    const float* Kv = branch ? &Outs[3][m][0] : &Outs[1][m][0];
    const float* Qv = &Outs[0][m][0];
    float acc = bb[o];
#pragma unroll 8
    for (int d = 0; d < DIM; ++d) acc += W[o * DIM + d] * (Kv[d] * Qv[d]);
    Hg[branch][m][o] = acc > 0.f ? acc : 0.f;
  }
  __syncthreads();

  if (t < 32) {
    int branch = t >> 4, m = t & 15;
    const float* W = branch ? Wf4 : Wf2;
    float acc = branch ? bf4[0] : bf2[0];
#pragma unroll 4
    for (int o = 0; o < H1DIM; ++o) acc += W[o] * Hg[branch][m][o];
    Sg[branch][m] = 1.f / (1.f + __expf(-acc));
  }
  __syncthreads();

  // ---- user = s1*pv + s2*nv ; emit uv = [user | items] as f16 ----
  for (int idx = t; idx < MT * DIM; idx += 128) {
    int m = idx >> 6, d = idx & 63;
    float user = Sg[0][m] * Outs[2][m][d] + Sg[1][m] * Outs[4][m][d];
    _Float16* dst = uvg + (size_t)(rowbase + m) * (2 * DIM);
    dst[d]       = (_Float16)user;
    dst[DIM + d] = (_Float16)Itm[m][d];
  }
}

// ---------------- Stage 3: fusion MLP 128 -> 1024 -> 512 -> 1 (WMMA) ----------
// 32-row tile, 2 M-subtiles of 16; each B fragment is reused for both subtiles.
// Dynamic LDS: uv (8KB) + H1 (64KB) + H2 (32KB) = 104KB (<320KB/WGP).

__global__ __launch_bounds__(256) void mlp_kernel(
    const _Float16* __restrict__ uvg,
    const _Float16* __restrict__ W1h, const float* __restrict__ b1,
    const _Float16* __restrict__ W2h, const float* __restrict__ b2,
    const float* __restrict__ W3,     const float* __restrict__ b3,
    float* __restrict__ out) {

  extern __shared__ __align__(32) char smem[];
  _Float16* A  = (_Float16*)smem;                 // MT2 x 128
  _Float16* H1 = A  + MT2 * 2 * DIM;              // MT2 x 1024
  _Float16* H2 = H1 + MT2 * H2DIM;                // MT2 x 512

  const int rowbase = blockIdx.x * MT2;
  const int wave = threadIdx.x >> 5, lane = threadIdx.x & 31;

  // stage uv tile into LDS (128-bit copies)
  {
    const uint4* src = (const uint4*)(uvg + (size_t)rowbase * 2 * DIM);
    uint4* dst = (uint4*)A;
    for (int idx = threadIdx.x; idx < MT2 * 2 * DIM / 8; idx += 256) dst[idx] = src[idx];
  }
  __syncthreads();

  // ---- layer 1: [32x128] @ [128x1024] ; 64 N-tiles x 2 M-subtiles ----
  {
    v16h am[2][4];
#pragma unroll
    for (int ms = 0; ms < 2; ++ms)
#pragma unroll
      for (int kc = 0; kc < 4; ++kc)
        am[ms][kc] = ldA(A + ms * 16 * 2 * DIM, kc * 32, 2 * DIM);
#pragma unroll
    for (int j = 0; j < 8; ++j) {
      const int nbase = (wave * 8 + j) * 16;
      v8f acc0 = {}, acc1 = {};
#pragma unroll
      for (int kc = 0; kc < 4; ++kc) {
        v16h bf = ldB(W1h, nbase, kc * 32, 128);   // loaded once, used twice
        acc0 = wmma32(am[0][kc], bf, acc0);
        acc1 = wmma32(am[1][kc], bf, acc1);
      }
      const int n  = nbase + (lane & 15);
      const int mo = (lane >> 4) * 8;
      const float bias = b1[n];
#pragma unroll
      for (int rr = 0; rr < 8; ++rr) {
        float v0 = acc0[rr] + bias;
        float v1 = acc1[rr] + bias;
        H1[(mo + rr) * H2DIM + n]        = (_Float16)(v0 > 0.f ? v0 : 0.f);
        H1[(16 + mo + rr) * H2DIM + n]   = (_Float16)(v1 > 0.f ? v1 : 0.f);
      }
    }
  }
  __syncthreads();

  // ---- layer 2: [32x1024] @ [1024x512] ; 32 N-tiles x 2 M-subtiles ----
  {
#pragma unroll
    for (int j = 0; j < 4; ++j) {
      const int nbase = (wave * 4 + j) * 16;
      v8f acc0 = {}, acc1 = {};
#pragma unroll 4
      for (int kc = 0; kc < 32; ++kc) {
        // prefetch next K-chunk of the B stream into cache (global_prefetch_b8)
        if (kc + 1 < 32) {
          const _Float16* pn = W2h + (size_t)(nbase + (lane & 15)) * H2DIM +
                               (kc + 1) * 32 + (lane >> 4) * 16;
          __builtin_prefetch(pn, 0, 1);
        }
        v16h bf  = ldB(W2h, nbase, kc * 32, H2DIM); // loaded once, used twice
        v16h af0 = ldA(H1,                 kc * 32, H2DIM);
        v16h af1 = ldA(H1 + 16 * H2DIM,    kc * 32, H2DIM);
        acc0 = wmma32(af0, bf, acc0);
        acc1 = wmma32(af1, bf, acc1);
      }
      const int n  = nbase + (lane & 15);
      const int mo = (lane >> 4) * 8;
      const float bias = b2[n];
#pragma unroll
      for (int rr = 0; rr < 8; ++rr) {
        float v0 = acc0[rr] + bias;
        float v1 = acc1[rr] + bias;
        H2[(mo + rr) * H3DIM + n]      = (_Float16)(v0 > 0.f ? v0 : 0.f);
        H2[(16 + mo + rr) * H3DIM + n] = (_Float16)(v1 > 0.f ? v1 : 0.f);
      }
    }
  }
  __syncthreads();

  // ---- layer 3: 512-dot + sigmoid per row ----
  if (threadIdx.x < MT2) {
    const int m = threadIdx.x;
    float acc = b3[0];
#pragma unroll 8
    for (int k = 0; k < H3DIM; ++k) acc += (float)H2[m * H3DIM + k] * W3[k];
    out[rowbase + m] = 1.f / (1.f + __expf(-acc));
  }
}

// ---------------- host launcher ----------------

extern "C" void kernel_launch(void* const* d_in, const int* in_sizes, int n_in,
                              void* d_out, int out_size, void* d_ws, size_t ws_size,
                              hipStream_t stream) {
  // input order (see setup_inputs):
  // 0 items_emb 1 users_df_emb 2 users_pos_emb 3 users_neg_emb
  // 4 Wq 5 bq 6 Wk1 7 bk1 8 Wv1 9 bv1 10 Wk2 11 bk2 12 Wv2 13 bv2
  // 14 Wf1 15 bf1 16 Wf2 17 bf2 18 Wf3 19 bf3 20 Wf4 21 bf4
  // 22 Wfc1 23 bfc1 24 Wfc2 25 bfc2 26 Wfc3 27 bfc3
  // 28 w2 29 w3 30 u 31 i 32 idx_G1 33 nb_G1 34 idx_G2 35 nb_G2 36 idx_G3 37 nb_G3
  (void)in_sizes; (void)n_in; (void)out_size; (void)ws_size;

  // workspace layout (f16 elements, all regions 32B aligned)
  _Float16* uvh  = (_Float16*)d_ws;                    // B * 128
  _Float16* WqH  = uvh  + (size_t)BTOT * 2 * DIM;      // 64*64
  _Float16* Wk1H = WqH  + 64 * 64;
  _Float16* Wv1H = Wk1H + 64 * 64;
  _Float16* Wk2H = Wv1H + 64 * 64;
  _Float16* Wv2H = Wk2H + 64 * 64;
  _Float16* W1h  = Wv2H + 64 * 64;                     // 1024*128
  _Float16* W2h  = W1h  + (size_t)H2DIM * 2 * DIM;     // 512*1024

  // convert weights to f16
  cvt_f32_f16<<<(64 * 64 + 255) / 256, 256, 0, stream>>>((const float*)d_in[4],  WqH,  64 * 64);
  cvt_f32_f16<<<(64 * 64 + 255) / 256, 256, 0, stream>>>((const float*)d_in[6],  Wk1H, 64 * 64);
  cvt_f32_f16<<<(64 * 64 + 255) / 256, 256, 0, stream>>>((const float*)d_in[8],  Wv1H, 64 * 64);
  cvt_f32_f16<<<(64 * 64 + 255) / 256, 256, 0, stream>>>((const float*)d_in[10], Wk2H, 64 * 64);
  cvt_f32_f16<<<(64 * 64 + 255) / 256, 256, 0, stream>>>((const float*)d_in[12], Wv2H, 64 * 64);
  cvt_f32_f16<<<(H2DIM * 2 * DIM + 255) / 256, 256, 0, stream>>>((const float*)d_in[22], W1h, H2DIM * 2 * DIM);
  cvt_f32_f16<<<(H3DIM * H2DIM + 255) / 256, 256, 0, stream>>>((const float*)d_in[24], W2h, H3DIM * H2DIM);

  // stage 2: gathers + QKV WMMA + gating -> uv f16
  stage2_kernel<<<BTOT / MT, 128, 0, stream>>>(
      (const float*)d_in[0], (const float*)d_in[1], (const float*)d_in[2], (const float*)d_in[3],
      WqH, Wk1H, Wv1H, Wk2H, Wv2H,
      (const float*)d_in[5], (const float*)d_in[7], (const float*)d_in[9],
      (const float*)d_in[11], (const float*)d_in[13],
      (const float*)d_in[14], (const float*)d_in[15], (const float*)d_in[16], (const float*)d_in[17],
      (const float*)d_in[18], (const float*)d_in[19], (const float*)d_in[20], (const float*)d_in[21],
      (const float*)d_in[28], (const float*)d_in[29],
      (const int*)d_in[31],
      (const int*)d_in[32], (const int*)d_in[33],
      (const int*)d_in[34], (const int*)d_in[35],
      (const int*)d_in[36], (const int*)d_in[37],
      uvh);

  // stage 3: fusion MLP (WMMA-heavy), 32-row tiles, dynamic LDS = 104KB
  const size_t smem_bytes =
      (size_t)(MT2 * 2 * DIM + MT2 * H2DIM + MT2 * H3DIM) * sizeof(_Float16);
  mlp_kernel<<<BTOT / MT2, 256, smem_bytes, stream>>>(
      uvh, W1h, (const float*)d_in[23], W2h, (const float*)d_in[25],
      (const float*)d_in[26], (const float*)d_in[27], (float*)d_out);
}
